// CostVolumeManager_47888885351073
// MI455X (gfx1250) — matrix operand order repaired and checked by reading, not compiled, not run on hardware
//
#include <hip/hip_runtime.h>
#include <stdint.h>

#ifndef __has_builtin
#define __has_builtin(x) 0
#endif
#if __has_builtin(__builtin_amdgcn_tensor_load_to_lds) && __has_builtin(__builtin_amdgcn_s_wait_tensorcnt)
#define CVM_USE_TDM 1
#else
#define CVM_USE_TDM 0
#endif

#define HH  64
#define WW  96
#define DDD 64
#define CC  16
#define VV  4
#define WT  (WW / 16)          // 6 tiles of 16 pixels per row
#define DCH 8                  // depth planes swept per wave
#define EPSF 1e-8f

typedef _Float16 v16h  __attribute__((ext_vector_type(16)));
typedef float    v8f   __attribute__((ext_vector_type(8)));
typedef uint32_t u32x4 __attribute__((ext_vector_type(4)));
typedef int      i32x4 __attribute__((ext_vector_type(4)));
typedef int      i32x8 __attribute__((ext_vector_type(8)));

// ---------------------------------------------------------------------------
// Prep kernel:
//   blocks 0..319 : transpose+convert one (view,y) row: (C,H,W) f32 -> (H,W,C) f16
//                   (view 4 == cur_feats). Strided channel rows staged into LDS
//                   via the Tensor Data Mover (tensor_load_to_lds, TENSORcnt).
//   block 320     : P = K @ E (rows 0..2) per view; inverse-depth plane table.
// ---------------------------------------------------------------------------
__global__ __launch_bounds__(256) void cvm_prep(
    const float* __restrict__ cur,    // (C,H,W)
    const float* __restrict__ src,    // (V,C,H,W)
    const float* __restrict__ E,      // (V,4,4)
    const float* __restrict__ Ks,     // (V,4,4)
    const float* __restrict__ minD,
    const float* __restrict__ maxD,
    _Float16* __restrict__ srcT,      // (V,H,W,C) f16
    _Float16* __restrict__ curT,      // (H,W,C)   f16
    float* __restrict__ Pw,           // (V,3,4)
    float* __restrict__ dplanes)      // (D)
{
    __shared__ float tile[CC * WW];   // 16 x 96 f32 = 6 KB
    const int b = blockIdx.x;
    const int t = threadIdx.x;

    if (b < (VV + 1) * HH) {
        const int v = b >> 6;         // 0..4 (4 == cur)
        const int y = b & 63;
        const float* sp = (v < VV) ? (src + (size_t)v * CC * HH * WW + (size_t)y * WW)
                                   : (cur + (size_t)y * WW);
#if CVM_USE_TDM
        if ((t >> 5) == 0) {          // one wave issues the TDM descriptor
            const uint32_t lds = (uint32_t)(uintptr_t)(&tile[0]);
            const uint64_t ga  = (uint64_t)(uintptr_t)sp;
            u32x4 g0;
            g0[0] = 1u;                                                // count=1, user mode
            g0[1] = lds;                                               // lds_addr
            g0[2] = (uint32_t)(ga & 0xFFFFFFFFu);                      // global_addr[31:0]
            g0[3] = (uint32_t)((ga >> 32) & 0x01FFFFFFu) | (2u << 30); // addr[56:32] | type=2
            i32x8 g1;
            g1[0] = (2 << 16);                 // data_size = 4 bytes
            g1[1] = (int)((HH * WW) << 16);    // tensor_dim0[15:0] = 6144
            g1[2] = (int)(CC << 16);           // tensor_dim0 hi=0 | tensor_dim1[15:0]=16
            g1[3] = (int)(WW << 16);           // tensor_dim1 hi=0 | tile_dim0=96
            g1[4] = CC;                        // tile_dim1=16, tile_dim2=0
            g1[5] = HH * WW;                   // tensor_dim0_stride = 6144 elements
            g1[6] = 0;
            g1[7] = 0;
            i32x4 gz = {0, 0, 0, 0};
#if __clang_major__ >= 23
            i32x8 gz8 = {0, 0, 0, 0, 0, 0, 0, 0};
            __builtin_amdgcn_tensor_load_to_lds(g0, g1, gz, gz, gz8, 0);
#else
            __builtin_amdgcn_tensor_load_to_lds(g0, g1, gz, gz, 0);
#endif
            __builtin_amdgcn_s_wait_tensorcnt(0);
        }
        __syncthreads();
#else
        for (int i = t; i < CC * WW; i += 256) {
            const int c = i / WW, x = i - c * WW;
            tile[i] = sp[(size_t)c * HH * WW + x];
        }
        __syncthreads();
#endif
        _Float16* dst = (v < VV) ? (srcT + (size_t)(v * HH + y) * WW * CC)
                                 : (curT + (size_t)y * WW * CC);
        for (int i = t; i < CC * WW; i += 256) {
            const int c = i / WW, x = i - c * WW;
            dst[(size_t)x * CC + c] = (_Float16)tile[c * WW + x];
        }
    } else {
        // P = (K @ E)[:3,:] per view
        if (t < VV * 12) {
            const int v = t / 12, r = t % 12, i = r >> 2, j = r & 3;
            const float* Kv = Ks + v * 16;
            const float* Ev = E  + v * 16;
            float s = 0.f;
            #pragma unroll
            for (int k = 0; k < 4; ++k) s += Kv[i * 4 + k] * Ev[k * 4 + j];
            Pw[v * 12 + i * 4 + j] = s;
        }
        // inverse-depth plane table
        if (t >= 64 && t < 64 + DDD) {
            const int dd = t - 64;
            const float im = 1.0f / minD[0], ix = 1.0f / maxD[0];
            const float r = (float)dd / (float)(DDD - 1);
            dplanes[dd] = 1.0f / (im + (ix - im) * r);
        }
    }
}

// Project one pixel into one source view at depth dp, bilinear-gather the
// 16-ch f16 feature vector (2 x b128 per corner), fold bilinear weights,
// border-zero padding and the z>0 mask into the blend.
__device__ __forceinline__ v16h cvm_gather_blend(
    const _Float16* __restrict__ base,   // this lane's view plane (H,W,C)
    float dp, float qx, float qy, float qz, float tx, float ty, float tz)
{
    const float cx = dp * qx + tx;
    const float cy = dp * qy + ty;
    const float cz = dp * qz + tz;
    const float zr = __builtin_amdgcn_rcpf(cz + EPSF);   // v_rcp_f32
    const float sx = cx * zr - 0.5f;
    const float sy = cy * zr - 0.5f;
    const float mz = (cz > 0.f) ? 1.f : 0.f;
    const float x0 = floorf(sx), y0 = floorf(sy);
    const float fx = sx - x0,    fy = sy - y0;
    const int  ix0 = (int)x0,    iy0 = (int)y0;

    v16h wa = {};
    #pragma unroll
    for (int k = 0; k < 4; ++k) {
        const int   xi  = ix0 + (k & 1);
        const int   yi  = iy0 + (k >> 1);
        const float wx  = (k & 1)  ? fx : (1.f - fx);
        const float wy  = (k >> 1) ? fy : (1.f - fy);
        const bool  ok  = (xi >= 0) & (xi < WW) & (yi >= 0) & (yi < HH);
        const float wgt = wx * wy * mz * (ok ? 1.f : 0.f);
        const int   xc  = xi < 0 ? 0 : (xi > WW - 1 ? WW - 1 : xi);
        const int   yc  = yi < 0 ? 0 : (yi > HH - 1 ? HH - 1 : yi);
        const u32x4* spp = (const u32x4*)(base + ((size_t)yc * WW + xc) * CC);
        union { v16h v; u32x4 q[2]; } Su;
        Su.q[0] = spp[0];
        Su.q[1] = spp[1];
        wa += Su.v * (_Float16)wgt;      // v_pk_fma_f16 x8
    }
    return wa;
}

// ---------------------------------------------------------------------------
// Main kernel: one wave32 per (depth-chunk, h, 16 consecutive w) tile; each
// wave sweeps DCH depth planes. Dense-K WMMA packing: the A fragment holds the
// cur-feature tile duplicated into both K halves (K0..15 == K16..31); the B
// fragment carries TWO views per WMMA (lanes 0..15 = view lo in K0..15,
// lanes 16..31 = view hi in K16..31), so each v_wmma_f32_16x16x32_f16
// accumulates two views' masked bilinear dots and two chained WMMAs cover all
// four views. Every lane gathers only its own view: no duplicate loads, no
// mask multiplies. The D-matrix diagonal is the 16 cost values (branchless
// cndmask select + one predicated store).
// ---------------------------------------------------------------------------
__global__ __launch_bounds__(256) void cvm_main(
    const _Float16* __restrict__ srcT,   // (V,H,W,C) f16
    const _Float16* __restrict__ curT,   // (H,W,C)   f16
    const float* __restrict__ Pw,        // (V,3,4)
    const float* __restrict__ dplanes,   // (D)
    const float* __restrict__ invK,      // (4,4)
    float* __restrict__ out)             // (D,H,W)
{
    const int wid     = blockIdx.x * 8 + (threadIdx.x >> 5);
    const int lane    = threadIdx.x & 31;
    const int n       = lane & 15;
    const int halfsel = lane >> 4;

    const int wt = wid % WT;
    const int h  = (wid / WT) % HH;
    const int dc = wid / (WT * HH);      // depth chunk 0..7
    const int w  = wt * 16 + n;

    // A fragment: lanes m and m+16 both hold cur ch0-15 (elements 0..15),
    // so each lane half contracts its own view: one WMMA contracts two views.
    union { v16h v; u32x4 q[2]; } Au;
    {
        const u32x4* curp = (const u32x4*)(curT + (size_t)(h * WW + w) * CC);
        Au.q[0] = curp[0];
        Au.q[1] = curp[1];
    }

    // This lane's two views: vA for WMMA#0, vB for WMMA#1.
    const int vA = halfsel;              // views {0,1} split across lane halves
    const int vB = 2 + halfsel;          // views {2,3}
    const _Float16* srcA = srcT + (size_t)vA * HH * WW * CC;
    const _Float16* srcB = srcT + (size_t)vB * HH * WW * CC;

    // Depth-invariant projection terms: cam_v(d) = dp * q_v + t_v
    const float px = (float)w + 0.5f, py = (float)h + 0.5f;
    const float rx = invK[0] * px + invK[1] * py + invK[2];
    const float ry = invK[4] * px + invK[5] * py + invK[6];
    const float rz = invK[8] * px + invK[9] * py + invK[10];

    const float* PA = Pw + vA * 12;
    const float* PB = Pw + vB * 12;
    const float qxA = PA[0] * rx + PA[1] * ry + PA[2]  * rz, txA = PA[3];
    const float qyA = PA[4] * rx + PA[5] * ry + PA[6]  * rz, tyA = PA[7];
    const float qzA = PA[8] * rx + PA[9] * ry + PA[10] * rz, tzA = PA[11];
    const float qxB = PB[0] * rx + PB[1] * ry + PB[2]  * rz, txB = PB[3];
    const float qyB = PB[4] * rx + PB[5] * ry + PB[6]  * rz, tyB = PB[7];
    const float qzB = PB[8] * rx + PB[9] * ry + PB[10] * rz, tzB = PB[11];

    const int j7   = lane & 7;
    const bool stl = (lane < 8) || (lane >= 24);       // lanes that store results
    const int  sn  = (lane < 16) ? lane : (lane - 16); // output index for those lanes

    for (int dd = 0; dd < DCH; ++dd) {
        const int d = dc * DCH + dd;
        const float dp = dplanes[d];

        v8f acc = {};

        // WMMA #0: views {0,1};  WMMA #1: views {2,3}  (chained accumulation)
        v16h b0 = cvm_gather_blend(srcA, dp, qxA, qyA, qzA, txA, tyA, tzA);
        acc = __builtin_amdgcn_wmma_f32_16x16x32_f16(
            false, Au.v, false, b0, (short)0, acc, false, false);

        v16h b1 = cvm_gather_blend(srcB, dp, qxB, qyB, qzB, txB, tyB, tzB);
        acc = __builtin_amdgcn_wmma_f32_16x16x32_f16(
            false, Au.v, false, b1, (short)0, acc, false, false);

        // Branchless diagonal extraction:
        //   diag[n] sits in (VGPR n, lane n) for n<8 and (VGPR n-8, lane n+16)
        //   for n>=8  =>  storing lanes need acc[lane & 7].
        float val = acc[0];
        #pragma unroll
        for (int j = 1; j < 8; ++j) val = (j7 == j) ? acc[j] : val;

        const size_t base = (size_t)(d * HH + h) * WW + (size_t)wt * 16;
        if (stl) out[base + sn] = val;
    }
}

// ---------------------------------------------------------------------------
extern "C" void kernel_launch(void* const* d_in, const int* in_sizes, int n_in,
                              void* d_out, int out_size, void* d_ws, size_t ws_size,
                              hipStream_t stream) {
    (void)in_sizes; (void)n_in; (void)out_size; (void)ws_size;

    const float* cur  = (const float*)d_in[0];   // (1,16,64,96)
    const float* src  = (const float*)d_in[1];   // (1,4,16,64,96)
    const float* E    = (const float*)d_in[2];   // (1,4,4,4)
    const float* Ks   = (const float*)d_in[3];   // (1,4,4,4)
    const float* invK = (const float*)d_in[4];   // (1,4,4)
    const float* mind = (const float*)d_in[5];
    const float* maxd = (const float*)d_in[6];

    char* ws = (char*)d_ws;
    _Float16* srcT    = (_Float16*)(ws);                      // 786432 B
    _Float16* curT    = (_Float16*)(ws + 786432);             // 196608 B
    float*    Pw      = (float*)   (ws + 786432 + 196608);    //    256 B
    float*    dplanes = (float*)   (ws + 786432 + 196608 + 256);

    cvm_prep<<<(VV + 1) * HH + 1, 256, 0, stream>>>(
        cur, src, E, Ks, mind, maxd, srcT, curT, Pw, dplanes);

    // 3072 wave-tiles: (D/DCH) depth chunks x H rows x 6 w-tiles, 8 waves/block
    cvm_main<<<((DDD / DCH) * HH * WT) / 8, 256, 0, stream>>>(
        srcT, curT, Pw, dplanes, invK, (float*)d_out);
}
